// MaskDINOEncoder_43739946942744
// MI455X (gfx1250) — matrix operand order repaired and checked by reading, not compiled, not run on hardware
//
#include <hip/hip_runtime.h>

#define USE_ASYNC_LDS 1

// ---------------- model constants ----------------
#define B_    4
#define L_    5440
#define D_    256
#define H_    8
#define DH_   32
#define DFF_  1024
#define NL_   6
#define M_    (B_ * L_)                 // 21760 rows (= 85 * 256)
#define MD_   ((size_t)M_ * D_)         // 5,570,560 elements

typedef unsigned short u16;
typedef __attribute__((ext_vector_type(16))) __bf16 v16bf;
typedef __attribute__((ext_vector_type(8)))  float  v8f;

union FragBF { v16bf v; uint4 q[2]; };

__device__ __forceinline__ u16 f2bf(float f) {
    unsigned u = __float_as_uint(f);
    unsigned r = u + 0x7FFFu + ((u >> 16) & 1u);   // round-to-nearest-even
    return (u16)(r >> 16);
}

// ---------------- elementwise kernels (vectorized x4) ----------------
__global__ __launch_bounds__(256) void copy_f32_kernel(const float* __restrict__ in,
                                                       float* __restrict__ out, int n4) {
    int i = blockIdx.x * 256 + threadIdx.x;
    if (i < n4) ((float4*)out)[i] = ((const float4*)in)[i];
}

__global__ __launch_bounds__(256) void lvlpos_kernel(const float* __restrict__ pos,
                                                     const float* __restrict__ lemb,
                                                     float* __restrict__ out) {
    int i = blockIdx.x * 256 + threadIdx.x;           // float4 index
    if (i >= (int)(MD_ / 4)) return;
    int e = i * 4;
    int tok = (e >> 8) % L_;
    int lev = (tok < 64) ? 0 : (tok < 320) ? 1 : (tok < 1344) ? 2 : 3;
    float4 p = ((const float4*)pos)[i];
    const float* le = lemb + lev * D_ + (e & 255);
    float4 o;
    o.x = p.x + le[0]; o.y = p.y + le[1]; o.z = p.z + le[2]; o.w = p.w + le[3];
    ((float4*)out)[i] = o;
}

__global__ __launch_bounds__(256) void cvtq_kernel(const float* __restrict__ x,
                                                   const float* __restrict__ lp,
                                                   u16* __restrict__ qb,
                                                   u16* __restrict__ xb) {
    int i = blockIdx.x * 256 + threadIdx.x;           // float4 index
    if (i >= (int)(MD_ / 4)) return;
    float4 xv = ((const float4*)x)[i];
    float4 lv = ((const float4*)lp)[i];
    ushort4 q, xo;
    q.x = f2bf(xv.x + lv.x); q.y = f2bf(xv.y + lv.y);
    q.z = f2bf(xv.z + lv.z); q.w = f2bf(xv.w + lv.w);
    xo.x = f2bf(xv.x); xo.y = f2bf(xv.y); xo.z = f2bf(xv.z); xo.w = f2bf(xv.w);
    ((ushort4*)qb)[i] = q;
    ((ushort4*)xb)[i] = xo;
}

// transpose + convert: in [K,N] f32 -> out [N,K] bf16
__global__ __launch_bounds__(256) void tcvt_kernel(const float* __restrict__ in,
                                                   u16* __restrict__ out, int K, int N) {
    int i = blockIdx.x * 256 + threadIdx.x;
    if (i >= K * N) return;
    int k = i / N, n = i - k * N;
    out[(size_t)n * K + k] = f2bf(in[i]);
}

// ---------------- WMMA GEMM ----------------
// C[M,N] = A[M,K](bf16) x Wt[N,K](bf16, pre-transposed) + bias
// block = 256 threads = 8 waves; wave tile = 32(M) x 64(N); block tile = 256 x 64.
// Weight chunk (64 cols x 256 k = 32 KB) staged in LDS via async-to-LDS copies.
__device__ __forceinline__ void loadA(FragBF& f, const u16* arow, int k, int hi) {
    f.q[0] = *(const uint4*)(arow + k + 8 * hi);        // K = k + 8*hi .. +7
    f.q[1] = *(const uint4*)(arow + k + 16 + 8 * hi);   // K = k + 16 + 8*hi .. +7
}

template <int RELU, int BF16OUT, int VLAYOUT>
__global__ __launch_bounds__(256) void gemm_kernel(const u16* __restrict__ A,
                                                   const u16* __restrict__ Wt,
                                                   const float* __restrict__ bias,
                                                   float* __restrict__ Cf,
                                                   u16* __restrict__ Cb,
                                                   int M, int N, int K) {
    __shared__ u16 ldsB[64 * 256];                      // 32 KB weight chunk
    const int lane = threadIdx.x & 31;
    const int wave = threadIdx.x >> 5;
    const int hi   = lane >> 4;
    const int ln16 = lane & 15;
    const int m0 = blockIdx.x * 256 + wave * 32;
    const int n0 = blockIdx.y * 64;

    v8f acc[2][4];
#pragma unroll
    for (int mi = 0; mi < 2; ++mi)
#pragma unroll
        for (int j = 0; j < 4; ++j)
#pragma unroll
            for (int r = 0; r < 8; ++r) acc[mi][j][r] = 0.0f;

    const u16* arow0 = A + (size_t)(m0 + ln16) * K;
    const u16* arow1 = A + (size_t)(m0 + 16 + ln16) * K;

    for (int kc = 0; kc < K; kc += 256) {
        // ---- cooperative stage of 64x256 weight chunk into LDS ----
        {
            int e = threadIdx.x;                        // uint4 entry; 2048 total, 8/thread
#pragma unroll
            for (int it = 0; it < 8; ++it, e += 256) {
                const int col = e >> 5;                 // 32 uint4 per column
                const int k8  = (e & 31) << 3;          // k offset (elements)
                const u16* srcp = Wt + (size_t)(n0 + col) * K + kc + k8;
#if USE_ASYNC_LDS
                unsigned ldst = (unsigned)(size_t)&ldsB[col * 256 + k8];
                unsigned long long ga = (unsigned long long)(size_t)srcp;
                asm volatile("global_load_async_to_lds_b128 %0, %1, off"
                             :: "v"(ldst), "v"(ga) : "memory");
#else
                *(uint4*)&ldsB[col * 256 + k8] = *(const uint4*)srcp;
#endif
            }
#if USE_ASYNC_LDS
            asm volatile("s_wait_asynccnt 0" ::: "memory");
#endif
            __syncthreads();
        }

        // ---- inner K loop over chunk, register-pipelined A ----
        FragBF a0c, a1c;
        loadA(a0c, arow0, kc, hi);
        loadA(a1c, arow1, kc, hi);
        for (int k = 0; k < 256; k += 32) {
            FragBF a0n, a1n;
            const bool pf = (kc + k + 32) < K;
            if (pf) {
                loadA(a0n, arow0, kc + k + 32, hi);
                loadA(a1n, arow1, kc + k + 32, hi);
            }
#pragma unroll
            for (int j = 0; j < 4; ++j) {
                FragBF b;
                const u16* bp = &ldsB[(j * 16 + ln16) * 256 + k + 8 * hi];
                b.q[0] = *(const uint4*)bp;
                b.q[1] = *(const uint4*)(bp + 16);
                acc[0][j] = __builtin_amdgcn_wmma_f32_16x16x32_bf16(
                    false, a0c.v, false, b.v, (short)0, acc[0][j], false, false);
                acc[1][j] = __builtin_amdgcn_wmma_f32_16x16x32_bf16(
                    false, a1c.v, false, b.v, (short)0, acc[1][j], false, false);
            }
            if (pf) { a0c = a0n; a1c = a1n; }
        }
        __syncthreads();                                // protect ldsB before restage
    }

#pragma unroll
    for (int mi = 0; mi < 2; ++mi) {
#pragma unroll
        for (int j = 0; j < 4; ++j) {
            const int col = n0 + j * 16 + ln16;
            const float bv = bias[col];
#pragma unroll
            for (int r = 0; r < 8; ++r) {
                const int row = m0 + mi * 16 + r + 8 * hi;
                float v = acc[mi][j][r] + bv;
                if (RELU) v = fmaxf(v, 0.0f);
                if (VLAYOUT) {
                    int b = row / L_, tok = row - b * L_;
                    int h = col >> 5, dh = col & 31;
                    Cf[((((size_t)b * H_ + h) * L_) + tok) * DH_ + dh] = v;
                } else if (BF16OUT) {
                    Cb[(size_t)row * N + col] = f2bf(v);
                } else {
                    Cf[(size_t)row * N + col] = v;
                }
            }
        }
    }
}

// ---------------- multi-scale deformable sampling ----------------
// one wave per (b, l, h); lane == dh channel (DH_ == 32 == wave32)
__global__ __launch_bounds__(256) void msda_kernel(const float* __restrict__ offb,   // [M,256]
                                                   const float* __restrict__ attnb,  // [M,128]
                                                   const float* __restrict__ vbuf,   // [B,H,L,32]
                                                   u16* __restrict__ sampb) {        // [M,256] bf16
    const int wid = blockIdx.x * 8 + (threadIdx.x >> 5);
    const int lane = threadIdx.x & 31;
    if (wid >= B_ * L_ * H_) return;
    const int h = wid & 7;
    const int bl = wid >> 3;
    const int l = bl % L_, b = bl / L_;

    int base, sz;
    if (l < 64)        { base = 0;    sz = 8;  }
    else if (l < 320)  { base = 64;   sz = 16; }
    else if (l < 1344) { base = 320;  sz = 32; }
    else               { base = 1344; sz = 64; }
    const int ll = l - base;
    const float rx = ((ll % sz) + 0.5f) / (float)sz;
    const float ry = ((ll / sz) + 0.5f) / (float)sz;

    const float* ap = attnb + (size_t)bl * 128 + h * 16;
    float lg[16];
    float mx = -1e30f;
#pragma unroll
    for (int i = 0; i < 16; ++i) { lg[i] = ap[i]; mx = fmaxf(mx, lg[i]); }
    float sum = 0.0f;
#pragma unroll
    for (int i = 0; i < 16; ++i) { lg[i] = __expf(lg[i] - mx); sum += lg[i]; }
    const float inv = 1.0f / sum;

    const float* op = offb + (size_t)bl * 256 + h * 32;
    const int levS[4] = {8, 16, 32, 64};
    const int levB[4] = {0, 64, 320, 1344};

    float o = 0.0f;
#pragma unroll
    for (int lv = 0; lv < 4; ++lv) {
        const int S = levS[lv];
        const float* vb = vbuf + ((((size_t)b * H_ + h) * L_) + levB[lv]) * DH_ + lane;
#pragma unroll
        for (int p = 0; p < 4; ++p) {
            const int ip = lv * 4 + p;
            const float x = (rx + op[ip * 2 + 0] / (float)S) * S - 0.5f;
            const float y = (ry + op[ip * 2 + 1] / (float)S) * S - 0.5f;
            const float fx0 = floorf(x), fy0 = floorf(y);
            const int x0 = (int)fx0, y0 = (int)fy0;
            const float fx = x - fx0, fy = y - fy0;
            float a4 = 0.0f;
            if (x0 >= 0 && x0 < S && y0 >= 0 && y0 < S)
                a4 += (1.0f - fx) * (1.0f - fy) * vb[(size_t)(y0 * S + x0) * DH_];
            if (x0 + 1 >= 0 && x0 + 1 < S && y0 >= 0 && y0 < S)
                a4 += fx * (1.0f - fy) * vb[(size_t)(y0 * S + x0 + 1) * DH_];
            if (x0 >= 0 && x0 < S && y0 + 1 >= 0 && y0 + 1 < S)
                a4 += (1.0f - fx) * fy * vb[(size_t)((y0 + 1) * S + x0) * DH_];
            if (x0 + 1 >= 0 && x0 + 1 < S && y0 + 1 >= 0 && y0 + 1 < S)
                a4 += fx * fy * vb[(size_t)((y0 + 1) * S + x0 + 1) * DH_];
            o += lg[ip] * inv * a4;
        }
    }
    sampb[(size_t)bl * 256 + h * 32 + lane] = f2bf(o);
}

// ---------------- residual add + LayerNorm (one row / block, D == 256) ----------------
__global__ __launch_bounds__(256) void add_ln_kernel(const float* __restrict__ xin,
                                                     const float* __restrict__ yin,
                                                     const float* __restrict__ g,
                                                     const float* __restrict__ bt,
                                                     float* __restrict__ xout,
                                                     u16* __restrict__ xbf) {
    __shared__ float sm[256];
    const int row = blockIdx.x, t = threadIdx.x;
    const size_t idx = (size_t)row * D_ + t;
    const float v = xin[idx] + yin[idx];

    sm[t] = v;
    __syncthreads();
    for (int s = 128; s > 0; s >>= 1) { if (t < s) sm[t] += sm[t + s]; __syncthreads(); }
    const float mean = sm[0] * (1.0f / D_);
    __syncthreads();
    const float d = v - mean;
    sm[t] = d * d;
    __syncthreads();
    for (int s = 128; s > 0; s >>= 1) { if (t < s) sm[t] += sm[t + s]; __syncthreads(); }
    const float var = sm[0] * (1.0f / D_);

    const float o = g[t] * d * rsqrtf(var + 1e-5f) + bt[t];
    xout[idx] = o;
    if (xbf) xbf[idx] = f2bf(o);
}

// ---------------- host launcher ----------------
extern "C" void kernel_launch(void* const* d_in, const int* in_sizes, int n_in,
                              void* d_out, int out_size, void* d_ws, size_t ws_size,
                              hipStream_t stream) {
    const float* src    = (const float*)d_in[0];
    const float* pos    = (const float*)d_in[1];
    const float* lemb   = (const float*)d_in[2];
    const float* w_off  = (const float*)d_in[3];
    const float* b_off  = (const float*)d_in[4];
    const float* w_attn = (const float*)d_in[5];
    const float* b_attn = (const float*)d_in[6];
    const float* w_val  = (const float*)d_in[7];
    const float* b_val  = (const float*)d_in[8];
    const float* w_out  = (const float*)d_in[9];
    const float* b_out  = (const float*)d_in[10];
    const float* ln1g   = (const float*)d_in[11];
    const float* ln1b   = (const float*)d_in[12];
    const float* w_ff1  = (const float*)d_in[13];
    const float* b_ff1  = (const float*)d_in[14];
    const float* w_ff2  = (const float*)d_in[15];
    const float* b_ff2  = (const float*)d_in[16];
    const float* ln2g   = (const float*)d_in[17];
    const float* ln2b   = (const float*)d_in[18];

    char* ws = (char*)d_ws;
    size_t cur = 0;
    auto alloc = [&](size_t bytes) -> char* {
        cur = (cur + 255) & ~(size_t)255;
        char* p = ws + cur;
        cur += bytes;
        return p;
    };
    float* x     = (float*)alloc(MD_ * 4);
    float* lp    = (float*)alloc(MD_ * 4);
    u16*   qb    = (u16*)  alloc(MD_ * 2);
    u16*   xb    = (u16*)  alloc(MD_ * 2);
    float* offb  = (float*)alloc(MD_ * 4);
    float* attnb = (float*)alloc((size_t)M_ * 128 * 4);
    float* vb    = (float*)alloc(MD_ * 4);
    float* sb    = (float*)alloc(MD_ * 4);
    u16*   ffh   = (u16*)  alloc((size_t)M_ * DFF_ * 2);
    const size_t WPER = 753664;  // bf16 weight elements per layer
    u16*   wts   = (u16*)  alloc((size_t)NL_ * WPER * 2);

    const int eg4 = (int)((MD_ / 4 + 255) / 256);
    copy_f32_kernel<<<eg4, 256, 0, stream>>>(src, x, (int)(MD_ / 4));
    lvlpos_kernel  <<<eg4, 256, 0, stream>>>(pos, lemb, lp);

    for (int i = 0; i < NL_; ++i) {
        u16* wl = wts + (size_t)i * WPER;
        u16 *valT = wl, *offT = wl + 65536, *attnT = wl + 131072,
            *outT = wl + 163840, *ff1T = wl + 229376, *ff2T = wl + 491520;
        tcvt_kernel<<<(65536 + 255) / 256, 256, 0, stream>>>(w_val + (size_t)i * 65536, valT, 256, 256);
        tcvt_kernel<<<(65536 + 255) / 256, 256, 0, stream>>>(w_off + (size_t)i * 65536, offT, 256, 256);
        tcvt_kernel<<<(32768 + 255) / 256, 256, 0, stream>>>(w_attn + (size_t)i * 32768, attnT, 256, 128);
        tcvt_kernel<<<(65536 + 255) / 256, 256, 0, stream>>>(w_out + (size_t)i * 65536, outT, 256, 256);
        tcvt_kernel<<<(262144 + 255) / 256, 256, 0, stream>>>(w_ff1 + (size_t)i * 262144, ff1T, 256, 1024);
        tcvt_kernel<<<(262144 + 255) / 256, 256, 0, stream>>>(w_ff2 + (size_t)i * 262144, ff2T, 1024, 256);
    }

    const int GM = M_ / 256;  // 85
    for (int i = 0; i < NL_; ++i) {
        u16* wl = wts + (size_t)i * WPER;
        u16 *valT = wl, *offT = wl + 65536, *attnT = wl + 131072,
            *outT = wl + 163840, *ff1T = wl + 229376, *ff2T = wl + 491520;

        cvtq_kernel<<<eg4, 256, 0, stream>>>(x, lp, qb, xb);

        gemm_kernel<0, 0, 1><<<dim3(GM, 4), 256, 0, stream>>>(
            xb, valT, b_val + (size_t)i * 256, vb, nullptr, M_, 256, 256);
        gemm_kernel<0, 0, 0><<<dim3(GM, 4), 256, 0, stream>>>(
            qb, offT, b_off + (size_t)i * 256, offb, nullptr, M_, 256, 256);
        gemm_kernel<0, 0, 0><<<dim3(GM, 2), 256, 0, stream>>>(
            qb, attnT, b_attn + (size_t)i * 128, attnb, nullptr, M_, 128, 256);

        msda_kernel<<<(B_ * L_ * H_) / 8, 256, 0, stream>>>(offb, attnb, vb, qb);

        gemm_kernel<0, 0, 0><<<dim3(GM, 4), 256, 0, stream>>>(
            qb, outT, b_out + (size_t)i * 256, sb, nullptr, M_, 256, 256);
        add_ln_kernel<<<M_, 256, 0, stream>>>(x, sb, ln1g + (size_t)i * 256, ln1b + (size_t)i * 256, x, xb);

        gemm_kernel<1, 1, 0><<<dim3(GM, 16), 256, 0, stream>>>(
            xb, ff1T, b_ff1 + (size_t)i * 1024, nullptr, ffh, M_, 1024, 256);
        gemm_kernel<0, 0, 0><<<dim3(GM, 4), 256, 0, stream>>>(
            ffh, ff2T, b_ff2 + (size_t)i * 256, sb, nullptr, M_, 256, 1024);
        add_ln_kernel<<<M_, 256, 0, stream>>>(x, sb, ln2g + (size_t)i * 256, ln2b + (size_t)i * 256, x, nullptr);
    }

    copy_f32_kernel<<<eg4, 256, 0, stream>>>(x, (float*)d_out, (int)(MD_ / 4));
}
